// TD3ActorRecurrentNetwork_23785528885442
// MI455X (gfx1250) — matrix operand order repaired and checked by reading, not compiled, not run on hardware
//
#include <hip/hip_runtime.h>
#include <hip/hip_bf16.h>

// ---------------------------------------------------------------------------
// TD3 recurrent actor forward, CDNA5 (gfx1250) WMMA bf16 path.
// B=8192, DE=256, DA=32, F=2048, H=2048, RATE=0.5
// Round 2: async global->LDS staging (ASYNCcnt) + 128x128 block tile.
// ---------------------------------------------------------------------------

#define B_SZ 8192
#define DE_SZ 256
#define DA_SZ 32
#define F_SZ  2048
#define H_SZ  2048

typedef __attribute__((ext_vector_type(16))) __bf16 v16bf;
typedef __attribute__((ext_vector_type(8)))  __bf16 v8bf;
typedef __attribute__((ext_vector_type(8)))  float  v8f;

union Frag16 { v16bf v; v8bf h[2]; };

// gfx1250 async copy: global -> LDS, 16B per lane, tracked by ASYNCcnt.
__device__ __forceinline__ void async_copy_b128(unsigned lds_off, const void* gptr)
{
    asm volatile("global_load_async_to_lds_b128 %0, %1, off"
                 :: "v"(lds_off), "v"(gptr) : "memory");
}
__device__ __forceinline__ void wait_async_le4()
{
    asm volatile("s_wait_asynccnt 0x4" ::: "memory");
}
__device__ __forceinline__ void wait_async_0()
{
    asm volatile("s_wait_asynccnt 0x0" ::: "memory");
}

// ---------------------------------------------------------------------------
// fp32 (strided 2D) -> bf16 (packed) conversion
// ---------------------------------------------------------------------------
__global__ void __launch_bounds__(256)
cvt_f32_bf16(const float* __restrict__ src, __hip_bfloat16* __restrict__ dst,
             int rows, int cols, int src_stride)
{
    size_t idx = (size_t)blockIdx.x * blockDim.x + threadIdx.x;
    size_t total = (size_t)rows * cols;
    if (idx >= total) return;
    size_t r = idx / cols;
    size_t c = idx - r * cols;
    dst[idx] = __float2bfloat16(src[r * (size_t)src_stride + c]);
}

// ---------------------------------------------------------------------------
// C = act(A @ W^T + bias)
//   A: M x K bf16 row-major (lda)
//   W: N x K bf16 row-major (ldw)   (WMMA B-operand = W^T, K x N)
// Block: 256 threads = 8 waves; block tile 128(M) x 128(N), K-step 32.
// Wave tile 32x64 = 2x4 of v_wmma_f32_16x16x32_bf16, acc[2][4].
// Staging: double-buffered LDS filled by global_load_async_to_lds_b128.
// Requires: M % 128 == 0, K % 32 == 0. N bounds-checked (clamped loads,
// guarded stores).
// ---------------------------------------------------------------------------
constexpr int MT = 128, NT = 128, KT = 32;
constexpr int SA = 40;   // padded LDS row stride in bf16 elems (80B, 16B-aligned)

__global__ void __launch_bounds__(256)
wmma_gemm(const __hip_bfloat16* __restrict__ A, int lda,
          const __hip_bfloat16* __restrict__ W, int ldw,
          const float* __restrict__ bias,
          float* __restrict__ Cf, __hip_bfloat16* __restrict__ Cb, int ldc,
          int N, int K, int actMode /*0=none,1=relu,2=tanh*/)
{
    __shared__ __align__(16) __hip_bfloat16 As[2][MT * SA];
    __shared__ __align__(16) __hip_bfloat16 Bs[2][NT * SA];

    const int tid  = threadIdx.x;
    const int wave = tid >> 5;
    const int lane = tid & 31;
    const int wm   = wave >> 1;          // 0..3 -> M sub-tile (32 rows each)
    const int wn   = wave & 1;           // 0..1 -> N sub-tile (64 cols each)
    const int m0   = blockIdx.y * MT;
    const int n0   = blockIdx.x * NT;
    const int half = lane >> 4;
    const int l16  = lane & 15;

    // staging task for this thread: 512 lane-tasks per tile, 2 per thread
    const int rowA0 = tid >> 2, quad = tid & 3;     // tasks tid and tid+256
    const int rowA1 = (tid + 256) >> 2;

    v8f acc[2][4] = {};

    const int nk = K / KT;

    auto stage = [&](int k, int buf) {
        const int k0 = k * KT;
        // A tile: 128 rows x 32 bf16 (4x16B per row)
        async_copy_b128((unsigned)(size_t)(&As[buf][rowA0 * SA + quad * 8]),
                        A + (size_t)(m0 + rowA0) * lda + k0 + quad * 8);
        async_copy_b128((unsigned)(size_t)(&As[buf][rowA1 * SA + quad * 8]),
                        A + (size_t)(m0 + rowA1) * lda + k0 + quad * 8);
        // B tile: 128 weight rows x 32 bf16 (clamped for N tail)
        int nb0 = n0 + rowA0; if (nb0 >= N) nb0 = N - 1;
        int nb1 = n0 + rowA1; if (nb1 >= N) nb1 = N - 1;
        async_copy_b128((unsigned)(size_t)(&Bs[buf][rowA0 * SA + quad * 8]),
                        W + (size_t)nb0 * ldw + k0 + quad * 8);
        async_copy_b128((unsigned)(size_t)(&Bs[buf][rowA1 * SA + quad * 8]),
                        W + (size_t)nb1 * ldw + k0 + quad * 8);
    };

    stage(0, 0);

    for (int k = 0; k < nk; ++k) {
        const int buf = k & 1;
        if (k + 1 < nk) {
            stage(k + 1, buf ^ 1);
            wait_async_le4();            // stage k fully landed (in-order done)
        } else {
            wait_async_0();
        }
        __syncthreads();                 // all waves' stage-k copies visible

        // --- fragments (ISA 7.12.2 16-bit layouts) ---
        Frag16 afrag[2], bfrag[4];
        #pragma unroll
        for (int i = 0; i < 2; ++i) {
            // A 16x32: half 0 -> K 0..7 & 16..23 ; half 1 -> K 8..15 & 24..31
            const __hip_bfloat16* ap = &As[buf][(wm * 32 + i * 16 + l16) * SA + half * 8];
            afrag[i].h[0] = *(const v8bf*)(ap);
            afrag[i].h[1] = *(const v8bf*)(ap + 16);
        }
        #pragma unroll
        for (int j = 0; j < 4; ++j) {
            // B 32x16 (= W rows): half 0 -> K 0..7 & 16..23 ; half 1 -> K 8..15 & 24..31
            const __hip_bfloat16* bp = &Bs[buf][(wn * 64 + j * 16 + l16) * SA + half * 8];
            bfrag[j].h[0] = *(const v8bf*)(bp);
            bfrag[j].h[1] = *(const v8bf*)(bp + 16);
        }

        #pragma unroll
        for (int i = 0; i < 2; ++i)
            #pragma unroll
            for (int j = 0; j < 4; ++j)
                acc[i][j] = __builtin_amdgcn_wmma_f32_16x16x32_bf16(
                    false, afrag[i].v, false, bfrag[j].v,
                    (short)0, acc[i][j], false, false);

        __syncthreads();                 // done reading buf before reuse
    }

    // --- epilogue: bias + activation + store (fp32 and/or bf16) ---
    #pragma unroll
    for (int i = 0; i < 2; ++i) {
        #pragma unroll
        for (int j = 0; j < 4; ++j) {
            int n = n0 + wn * 64 + j * 16 + l16;
            if (n >= N) continue;
            float bv = bias ? bias[n] : 0.0f;
            #pragma unroll
            for (int r = 0; r < 8; ++r) {
                int m = m0 + wm * 32 + i * 16 + half * 8 + r;
                float v = acc[i][j][r] + bv;
                if (actMode == 1)      v = fmaxf(v, 0.0f);
                else if (actMode == 2) v = tanhf(v);
                if (Cf) Cf[(size_t)m * ldc + n] = v;
                if (Cb) Cb[(size_t)m * ldc + n] = __float2bfloat16(v);
            }
        }
    }
}

// ---------------------------------------------------------------------------
// GRU gating + next_hidden + h1(bf16) packed into [feat_post | h1] buffer
// ---------------------------------------------------------------------------
__global__ void __launch_bounds__(256)
gru_kernel(const float* __restrict__ gi, const float* __restrict__ gh,
           const float* __restrict__ state,
           float* __restrict__ next_hidden, __hip_bfloat16* __restrict__ fpbuf)
{
    size_t idx = (size_t)blockIdx.x * blockDim.x + threadIdx.x;
    if (idx >= (size_t)B_SZ * H_SZ) return;
    size_t b = idx / H_SZ;
    int    h = (int)(idx - b * H_SZ);

    const float* gib = gi + b * (size_t)(3 * H_SZ);
    const float* ghb = gh + b * (size_t)(3 * H_SZ);
    float ir = gib[h], iz = gib[H_SZ + h], in_ = gib[2 * H_SZ + h];
    float hr = ghb[h], hz = ghb[H_SZ + h], hn  = ghb[2 * H_SZ + h];
    float h0 = state[b * (size_t)(DE_SZ + H_SZ) + DE_SZ + h];

    float r = 1.0f / (1.0f + __expf(-(ir + hr)));
    float z = 1.0f / (1.0f + __expf(-(iz + hz)));
    float n = tanhf(in_ + r * hn);
    float h1 = (1.0f - z) * n + z * h0;

    next_hidden[idx] = 0.5f * h0 + 0.5f * h1;                     // RATE = 0.5
    fpbuf[b * (size_t)(2 * F_SZ) + F_SZ + h] = __float2bfloat16(h1);
}

// ---------------------------------------------------------------------------
// launch
// ---------------------------------------------------------------------------
static inline size_t align256(size_t x) { return (x + 255) & ~(size_t)255; }

extern "C" void kernel_launch(void* const* d_in, const int* in_sizes, int n_in,
                              void* d_out, int out_size, void* d_ws, size_t ws_size,
                              hipStream_t stream)
{
    const float* state  = (const float*)d_in[0];
    const float* p_act  = (const float*)d_in[1];
    const float* W_o    = (const float*)d_in[2];
    const float* b_o    = (const float*)d_in[3];
    const float* W_a    = (const float*)d_in[4];
    const float* b_a    = (const float*)d_in[5];
    const float* W_post = (const float*)d_in[6];
    const float* b_post = (const float*)d_in[7];
    const float* W_ih   = (const float*)d_in[8];
    const float* W_hh   = (const float*)d_in[9];
    const float* b_ih   = (const float*)d_in[10];
    const float* b_hh   = (const float*)d_in[11];
    const float* W3     = (const float*)d_in[12];
    const float* b3     = (const float*)d_in[13];

    float* out_action = (float*)d_out;                          // B x DA
    float* out_nh     = (float*)d_out + (size_t)B_SZ * DA_SZ;   // B x H

    // ---- workspace carve ----
    char* ws = (char*)d_ws;  size_t off = 0;
    auto carve = [&](size_t bytes) { char* p = ws + off; off = align256(off + bytes); return p; };
    __hip_bfloat16* env_bf  = (__hip_bfloat16*)carve((size_t)B_SZ * DE_SZ * 2);
    __hip_bfloat16* act_bf  = (__hip_bfloat16*)carve((size_t)B_SZ * DA_SZ * 2);
    __hip_bfloat16* h0_bf   = (__hip_bfloat16*)carve((size_t)B_SZ * H_SZ * 2);
    __hip_bfloat16* x_bf    = (__hip_bfloat16*)carve((size_t)B_SZ * 2 * F_SZ * 2);   // [feat_o|feat_a]
    __hip_bfloat16* fp_bf   = (__hip_bfloat16*)carve((size_t)B_SZ * 2 * F_SZ * 2);   // [feat_post|h1]
    float*          gi      = (float*)carve((size_t)B_SZ * 3 * H_SZ * 4);
    float*          gh      = (float*)carve((size_t)B_SZ * 3 * H_SZ * 4);
    __hip_bfloat16* Wo_bf   = (__hip_bfloat16*)carve((size_t)F_SZ * DE_SZ * 2);
    __hip_bfloat16* Wa_bf   = (__hip_bfloat16*)carve((size_t)F_SZ * DA_SZ * 2);
    __hip_bfloat16* Wp_bf   = (__hip_bfloat16*)carve((size_t)F_SZ * DE_SZ * 2);
    __hip_bfloat16* Wih_bf  = (__hip_bfloat16*)carve((size_t)3 * H_SZ * 2 * F_SZ * 2);
    __hip_bfloat16* Whh_bf  = (__hip_bfloat16*)carve((size_t)3 * H_SZ * H_SZ * 2);
    __hip_bfloat16* W3_bf   = (__hip_bfloat16*)carve((size_t)DA_SZ * (F_SZ + H_SZ) * 2);

    auto cvt = [&](const float* src, __hip_bfloat16* dst, int rows, int cols, int stride) {
        size_t total = (size_t)rows * cols;
        int blocks = (int)((total + 255) / 256);
        hipLaunchKernelGGL(cvt_f32_bf16, dim3(blocks), dim3(256), 0, stream,
                           src, dst, rows, cols, stride);
    };

    // ---- 0) fp32 -> bf16 conversions ----
    cvt(state,            env_bf, B_SZ, DE_SZ, DE_SZ + H_SZ);
    cvt(state + DE_SZ,    h0_bf,  B_SZ, H_SZ,  DE_SZ + H_SZ);
    cvt(p_act,            act_bf, B_SZ, DA_SZ, DA_SZ);
    cvt(W_o,    Wo_bf,  F_SZ,     DE_SZ,        DE_SZ);
    cvt(W_a,    Wa_bf,  F_SZ,     DA_SZ,        DA_SZ);
    cvt(W_post, Wp_bf,  F_SZ,     DE_SZ,        DE_SZ);
    cvt(W_ih,   Wih_bf, 3 * H_SZ, 2 * F_SZ,     2 * F_SZ);
    cvt(W_hh,   Whh_bf, 3 * H_SZ, H_SZ,         H_SZ);
    cvt(W3,     W3_bf,  DA_SZ,    F_SZ + H_SZ,  F_SZ + H_SZ);

    auto gemm = [&](const __hip_bfloat16* A, int lda,
                    const __hip_bfloat16* Wm, int ldw, const float* bias,
                    float* Cf, __hip_bfloat16* Cb, int ldc,
                    int M, int N, int K, int actMode) {
        dim3 grid((N + NT - 1) / NT, M / MT);
        hipLaunchKernelGGL(wmma_gemm, grid, dim3(256), 0, stream,
                           A, lda, Wm, ldw, bias, Cf, Cb, ldc, N, K, actMode);
    };

    // ---- 1) feat_o = relu(env @ W_o^T + b_o)  -> x[:, :F] (bf16) ----
    gemm(env_bf, DE_SZ, Wo_bf, DE_SZ, b_o, nullptr, x_bf, 2 * F_SZ,
         B_SZ, F_SZ, DE_SZ, 1);
    // ---- 2) feat_a = relu(prev_action @ W_a^T + b_a) -> x[:, F:2F] ----
    gemm(act_bf, DA_SZ, Wa_bf, DA_SZ, b_a, nullptr, x_bf + F_SZ, 2 * F_SZ,
         B_SZ, F_SZ, DA_SZ, 1);
    // ---- 3) feat_post = relu(env @ W_post^T + b_post) -> fp[:, :F] ----
    gemm(env_bf, DE_SZ, Wp_bf, DE_SZ, b_post, nullptr, fp_bf, 2 * F_SZ,
         B_SZ, F_SZ, DE_SZ, 1);
    // ---- 4) gi = x @ W_ih^T + b_ih  (fp32) ----
    gemm(x_bf, 2 * F_SZ, Wih_bf, 2 * F_SZ, b_ih, gi, nullptr, 3 * H_SZ,
         B_SZ, 3 * H_SZ, 2 * F_SZ, 0);
    // ---- 5) gh = h0 @ W_hh^T + b_hh (fp32) ----
    gemm(h0_bf, H_SZ, Whh_bf, H_SZ, b_hh, gh, nullptr, 3 * H_SZ,
         B_SZ, 3 * H_SZ, H_SZ, 0);

    // ---- 6) GRU gating: next_hidden -> d_out, h1(bf16) -> fp[:, F:2F] ----
    {
        size_t total = (size_t)B_SZ * H_SZ;
        int blocks = (int)((total + 255) / 256);
        hipLaunchKernelGGL(gru_kernel, dim3(blocks), dim3(256), 0, stream,
                           gi, gh, state, out_nh, fp_bf);
    }

    // ---- 7) action = tanh([feat_post|h1] @ W3^T + b3) -> d_out ----
    gemm(fp_bf, 2 * F_SZ, W3_bf, F_SZ + H_SZ, b3, out_action, nullptr, DA_SZ,
         B_SZ, DA_SZ, F_SZ + H_SZ, 2);

    (void)in_sizes; (void)n_in; (void)out_size; (void)ws_size;
}
